// MultiHeadAttention_72799695667250
// MI455X (gfx1250) — compile-verified, matrix operand order checked by reading, not run on hardware
//
#include <hip/hip_runtime.h>
#include <hip/hip_bf16.h>
#include <math.h>

#define BATCH 4
#define SEQ   1024
#define DIM   1024
#define NHEAD 16
#define DHEAD 64
#define KW    3
#define NBH   (BATCH * NHEAD)   // 64 (flat-order reshape groups)

#define PANEL_ROWS 32           // B-panel rows staged in LDS per stage
#define PANEL_K    512          // K extent per stage
#define LDSROW     520          // elements per LDS row: 512 + 8 (TDM pad: 4 dwords / 256 dwords)

typedef __attribute__((ext_vector_type(16))) __bf16       v16bf;
typedef __attribute__((ext_vector_type(8)))  float        v8f;
typedef __attribute__((ext_vector_type(4)))  unsigned int u32x4;
typedef __attribute__((ext_vector_type(8)))  int          i32x8;
typedef __attribute__((ext_vector_type(4)))  int          i32x4;

union ABFrag { v16bf bf; u32x4 q[2]; };

__device__ __forceinline__ unsigned short f2bf(float f) {
  unsigned int u = __float_as_uint(f);
  unsigned int r = u + 0x7FFFu + ((u >> 16) & 1u);   // round-to-nearest-even
  return (unsigned short)(r >> 16);
}

// WMMA A/B fragment load (16-bit, 16xK tile): lane = 16*half + r holds row r;
// q[0] = K[8*half .. +7], q[1] = K[16+8*half .. +7].
__device__ __forceinline__ void load_frag(ABFrag& f, const unsigned short* rowk, int half) {
  f.q[0] = *(const u32x4*)(rowk + 8 * half);
  f.q[1] = *(const u32x4*)(rowk + 16 + 8 * half);
}

__device__ __forceinline__ v8f wmma_bf16(const ABFrag& a, const ABFrag& b, v8f c) {
  return __builtin_amdgcn_wmma_f32_16x16x32_bf16(false, a.bf, false, b.bf,
                                                 (short)0, c, false, false);
}

// ---------------------------------------------------------------------------
// Tensor Data Mover: DMA a 2D tile (tile_x elems wide, tile_y rows, element =
// 2 bytes, global row stride = stride_x elems) into LDS at lds_off, inserting
// 4 dwords of padding every 256 dwords (-> LDS row stride = tile_x + 8 elems
// when tile_x == 512), for bank-conflict-free ds_load_b128 of rows.
// D# layout per CDNA5 ISA.
// ---------------------------------------------------------------------------
__device__ __forceinline__ void tdm_load_2d(unsigned lds_off, const void* gaddr,
                                            unsigned tile_x, unsigned tile_y,
                                            unsigned stride_x) {
  unsigned long long ga = (unsigned long long)(uintptr_t)gaddr;
  u32x4 g0;
  g0[0] = 1u;                                        // count=1 (valid), user mode
  g0[1] = lds_off;                                   // lds_addr (bytes)
  g0[2] = (unsigned)(ga & 0xFFFFFFFFull);            // global_addr[31:0]
  g0[3] = (unsigned)((ga >> 32) & 0x01FFFFFFull)     // global_addr[56:32]
          | 0x80000000u;                             // type = 2 ("image")
  i32x8 g1;
  // [15:0] wg_mask=0 | [17:16] data_size=1 (2B) | [20] pad_en=1
  // | [24:22] pad_interval=7 (256 dwords) | [31:25] pad_amount=3 (4 dwords)
  g1[0] = (int)0x07D10000;
  g1[1] = (int)((tile_x & 0xFFFFu) << 16);                       // tensor_dim0[15:0]
  g1[2] = (int)(((tile_x >> 16) & 0xFFFFu) | ((tile_y & 0xFFFFu) << 16)); // dim0 hi | dim1 lo
  g1[3] = (int)(((tile_y >> 16) & 0xFFFFu) | ((tile_x & 0xFFFFu) << 16)); // dim1 hi | tile_dim0
  g1[4] = (int)(tile_y & 0xFFFFu);                   // tile_dim1 | tile_dim2=0
  g1[5] = (int)stride_x;                             // tensor_dim0_stride[31:0]
  g1[6] = 0;                                         // stride hi | dim1_stride lo
  g1[7] = 0;
  i32x4 z4 = {0, 0, 0, 0};
#if defined(__clang_major__) && (__clang_major__ >= 23)
  i32x8 z8 = {0, 0, 0, 0, 0, 0, 0, 0};
  __builtin_amdgcn_tensor_load_to_lds(g0, g1, z4, z4, z8, 0);
#else
  __builtin_amdgcn_tensor_load_to_lds(g0, g1, z4, z4, 0);
#endif
}

// ---------------------------------------------------------------- converts
__global__ void __launch_bounds__(256) k_cvt_bf16(const float* __restrict__ in,
                                                  unsigned short* __restrict__ out, int n) {
  int i = blockIdx.x * 256 + threadIdx.x;
  if (i < n) out[i] = f2bf(in[i]);
}

// w[o][i][j] (D,D,3) -> out[j][o][i] (3,D,D)
__global__ void __launch_bounds__(256) k_cvt_convw(const float* __restrict__ w,
                                                   unsigned short* __restrict__ out) {
  int oi = blockIdx.x * 256 + threadIdx.x;
  if (oi >= DIM * DIM) return;
#pragma unroll
  for (int j = 0; j < KW; ++j) out[(size_t)j * DIM * DIM + oi] = f2bf(w[(size_t)oi * KW + j]);
}

// --------------------------------------------------- conv-as-GEMM (+GELU)
// 256 threads = 8 waves; wave w computes rows [blk*128 + 16w, +16).
// B-panel (32 output channels x 512 K) TDM-staged in LDS, reused by all 8
// waves; 3 taps x 2 K-halves = 6 stages accumulated into the same C.
__global__ void __launch_bounds__(256) k_conv_gemm(
    const unsigned short* __restrict__ xbf,   // (B,L,D) bf16
    const unsigned short* __restrict__ wbf,   // (3,D,D) bf16  [tap][o][i]
    const float* __restrict__ bias,
    unsigned short* __restrict__ ybf)         // (B,L,D) bf16
{
  __shared__ __align__(128) unsigned short Bpan[PANEL_ROWS * LDSROW];  // 33280 B

  const int lane = threadIdx.x & 31;
  const int wave = threadIdx.x >> 5;          // 0..7
  const int r = lane & 15, half = lane >> 4;
  const int grow = blockIdx.x * 128 + wave * 16;   // row in (B*L); never crosses batch
  const int b  = grow / SEQ;
  const int t0 = grow % SEQ;
  const int n0 = blockIdx.y * PANEL_ROWS;
  const unsigned short* X = xbf + (size_t)b * SEQ * DIM;
  const unsigned lds_off = (unsigned)(uintptr_t)(void*)&Bpan[0];

  v8f c[2] = {};
#pragma unroll
  for (int j = 0; j < KW; ++j) {
    const int row = (t0 + r + j - 2 + SEQ) % SEQ;
    const unsigned short* Arow = X + (size_t)row * DIM;
#pragma unroll
    for (int kh = 0; kh < DIM / PANEL_K; ++kh) {
      __syncthreads();                        // previous panel fully consumed
      if (wave == 0) {
        tdm_load_2d(lds_off,
                    wbf + (size_t)j * DIM * DIM + (size_t)n0 * DIM + kh * PANEL_K,
                    PANEL_K, PANEL_ROWS, DIM);
        __builtin_amdgcn_s_wait_tensorcnt(0);
      }
      __syncthreads();                        // panel ready
      for (int kc = 0; kc < PANEL_K; kc += 32) {
        ABFrag a; load_frag(a, Arow + kh * PANEL_K + kc, half);
#pragma unroll
        for (int nt = 0; nt < 2; ++nt) {
          ABFrag bm; load_frag(bm, &Bpan[(nt * 16 + r) * LDSROW + kc], half);
          c[nt] = wmma_bf16(a, bm, c[nt]);
        }
      }
    }
  }
#pragma unroll
  for (int nt = 0; nt < 2; ++nt) {
    const int col = n0 + nt * 16 + r;
    const float bcol = bias[col];
#pragma unroll
    for (int i = 0; i < 8; ++i) {
      const int m = i + 8 * half;             // C layout: VGPR i -> row i+8*half
      float x = c[nt][i] + bcol;
      float g = 0.5f * x * (1.0f + erff(x * 0.70710678118654752f));
      ybf[((size_t)b * SEQ + t0 + m) * DIM + col] = f2bf(g);
    }
  }
}

// ----------------------------------------------- QK^T -> log(clamp(s*0.5))
__global__ void __launch_bounds__(32) k_qk_logits(
    const unsigned short* __restrict__ qbf,   // flat (64, L, 64) bf16
    const unsigned short* __restrict__ kbf,
    float* __restrict__ attn)                 // (64, L, L) f32
{
  const int lane = threadIdx.x & 31;
  const int r = lane & 15, half = lane >> 4;
  const int bh = blockIdx.x;
  const int t0 = blockIdx.y * 16;
  const int s0 = blockIdx.z * 64;
  const unsigned short* Q = qbf + (size_t)bh * SEQ * DHEAD;
  const unsigned short* K = kbf + (size_t)bh * SEQ * DHEAD;

  v8f c[4] = {};
  for (int kc = 0; kc < DHEAD; kc += 32) {
    ABFrag a; load_frag(a, Q + (size_t)(t0 + r) * DHEAD + kc, half);
#pragma unroll
    for (int nt = 0; nt < 4; ++nt) {
      ABFrag bm; load_frag(bm, K + (size_t)(s0 + nt * 16 + r) * DHEAD + kc, half);
      c[nt] = wmma_bf16(a, bm, c[nt]);
    }
  }
  float* out = attn + (size_t)bh * SEQ * SEQ;
#pragma unroll
  for (int nt = 0; nt < 4; ++nt) {
    const int scol = s0 + nt * 16 + r;
#pragma unroll
    for (int i = 0; i < 8; ++i) {
      const int m = i + 8 * half;
      float v = c[nt][i] * 0.5f;              // scale = (d//H)^-0.5 = 0.5
      v = (v < 0.1f) ? 0.1f : v;
      out[(size_t)(t0 + m) * SEQ + scol] = logf(v);
    }
  }
}

// -------------------------------------------- per-row max + logsumexp
__global__ void __launch_bounds__(256) k_row_stats(const float* __restrict__ attn,
                                                   float* __restrict__ logZ) {
  __shared__ float sdata[256];
  const float* p = attn + (size_t)blockIdx.x * SEQ;
  float v[4];
#pragma unroll
  for (int i = 0; i < 4; ++i) v[i] = p[threadIdx.x + 256 * i];
  float m = fmaxf(fmaxf(v[0], v[1]), fmaxf(v[2], v[3]));
  sdata[threadIdx.x] = m; __syncthreads();
  for (int s = 128; s > 0; s >>= 1) {
    if (threadIdx.x < s) sdata[threadIdx.x] = fmaxf(sdata[threadIdx.x], sdata[threadIdx.x + s]);
    __syncthreads();
  }
  const float M = sdata[0]; __syncthreads();
  float sum = 0.f;
#pragma unroll
  for (int i = 0; i < 4; ++i) sum += expf(v[i] - M);
  sdata[threadIdx.x] = sum; __syncthreads();
  for (int s = 128; s > 0; s >>= 1) {
    if (threadIdx.x < s) sdata[threadIdx.x] += sdata[threadIdx.x + s];
    __syncthreads();
  }
  if (threadIdx.x == 0) logZ[blockIdx.x] = M + logf(sdata[0]);
}

// -------- finalize log-softmax: write f32 result + bf16 copy for GEMM
__global__ void __launch_bounds__(256) k_apply_lsm(float* __restrict__ attn,
                                                   const float* __restrict__ logZ,
                                                   unsigned short* __restrict__ attnbf) {
  size_t idx = (size_t)blockIdx.x * 256 + threadIdx.x;
  int row = (int)(idx >> 10);
  float v = attn[idx] - logZ[row];
  attn[idx] = v;
  attnbf[idx] = f2bf(v);
}

// -------- v (64,L,64) -> vT (64,64,L) so context GEMM B is K-contiguous
__global__ void __launch_bounds__(256) k_transpose_v(const unsigned short* __restrict__ vbf,
                                                     unsigned short* __restrict__ vT) {
  int idx = blockIdx.x * 256 + threadIdx.x;     // B*L*D elements
  int bh = idx >> 16, rem = idx & 65535, s = rem >> 6, dd = rem & 63;
  vT[((size_t)bh * DHEAD + dd) * SEQ + s] = vbf[idx];
}

// ------------------------------- context = attn(log-probs) @ V
__global__ void __launch_bounds__(32) k_context_gemm(
    const unsigned short* __restrict__ attnbf,  // (64,L,L)
    const unsigned short* __restrict__ vT,      // (64,64,L)
    unsigned short* __restrict__ ctxbf)         // flat (B*L, D)
{
  const int lane = threadIdx.x & 31;
  const int r = lane & 15, half = lane >> 4;
  const int bh = blockIdx.x;
  const int t0 = blockIdx.y * 16;
  const unsigned short* A  = attnbf + (size_t)bh * SEQ * SEQ;
  const unsigned short* Bm = vT + (size_t)bh * DHEAD * SEQ;

  v8f c[4] = {};
  for (int kc = 0; kc < SEQ; kc += 32) {
    ABFrag a; load_frag(a, A + (size_t)(t0 + r) * SEQ + kc, half);
#pragma unroll
    for (int nt = 0; nt < 4; ++nt) {
      ABFrag bm; load_frag(bm, Bm + (size_t)(nt * 16 + r) * SEQ + kc, half);
      c[nt] = wmma_bf16(a, bm, c[nt]);
    }
  }
#pragma unroll
  for (int nt = 0; nt < 4; ++nt) {
    const int dd = nt * 16 + r;
#pragma unroll
    for (int i = 0; i < 8; ++i) {
      const int m = i + 8 * half;
      ctxbf[((size_t)bh * SEQ + t0 + m) * DHEAD + dd] = f2bf(c[nt][i]);
    }
  }
}

// ------------------- output = ctx @ wf^T + bf + residual (pre-LayerNorm)
// Same TDM-staged B-panel scheme as the conv GEMM (1 "tap").
__global__ void __launch_bounds__(256) k_proj_gemm(
    const unsigned short* __restrict__ ctxbf,   // (B*L, D)
    const unsigned short* __restrict__ wfbf,    // (D, D) [o][d]
    const float* __restrict__ bias,
    const float* __restrict__ residual,         // query f32
    float* __restrict__ preln)
{
  __shared__ __align__(128) unsigned short Bpan[PANEL_ROWS * LDSROW];

  const int lane = threadIdx.x & 31;
  const int wave = threadIdx.x >> 5;
  const int r = lane & 15, half = lane >> 4;
  const int row0 = blockIdx.x * 128 + wave * 16;
  const int n0 = blockIdx.y * PANEL_ROWS;
  const unsigned lds_off = (unsigned)(uintptr_t)(void*)&Bpan[0];
  const unsigned short* Arow = ctxbf + (size_t)(row0 + r) * DIM;

  v8f c[2] = {};
#pragma unroll
  for (int kh = 0; kh < DIM / PANEL_K; ++kh) {
    __syncthreads();
    if (wave == 0) {
      tdm_load_2d(lds_off, wfbf + (size_t)n0 * DIM + kh * PANEL_K,
                  PANEL_K, PANEL_ROWS, DIM);
      __builtin_amdgcn_s_wait_tensorcnt(0);
    }
    __syncthreads();
    for (int kc = 0; kc < PANEL_K; kc += 32) {
      ABFrag a; load_frag(a, Arow + kh * PANEL_K + kc, half);
#pragma unroll
      for (int nt = 0; nt < 2; ++nt) {
        ABFrag bm; load_frag(bm, &Bpan[(nt * 16 + r) * LDSROW + kc], half);
        c[nt] = wmma_bf16(a, bm, c[nt]);
      }
    }
  }
#pragma unroll
  for (int nt = 0; nt < 2; ++nt) {
    const int col = n0 + nt * 16 + r;
    const float bc = bias[col];
#pragma unroll
    for (int i = 0; i < 8; ++i) {
      const int m = i + 8 * half;
      size_t off = (size_t)(row0 + m) * DIM + col;
      preln[off] = c[nt][i] + bc + residual[off];
    }
  }
}

// ---------------------------------------------------------- LayerNorm
__global__ void __launch_bounds__(256) k_layernorm(const float* __restrict__ x,
                                                   const float* __restrict__ g,
                                                   const float* __restrict__ b,
                                                   float* __restrict__ out) {
  __shared__ float sdata[256];
  const float* p = x + (size_t)blockIdx.x * DIM;
  float v[4];
#pragma unroll
  for (int i = 0; i < 4; ++i) v[i] = p[threadIdx.x + 256 * i];
  sdata[threadIdx.x] = v[0] + v[1] + v[2] + v[3]; __syncthreads();
  for (int s = 128; s > 0; s >>= 1) {
    if (threadIdx.x < s) sdata[threadIdx.x] += sdata[threadIdx.x + s];
    __syncthreads();
  }
  float mu = sdata[0] * (1.0f / DIM); __syncthreads();
  float sq = 0.f;
#pragma unroll
  for (int i = 0; i < 4; ++i) { float d = v[i] - mu; sq += d * d; }
  sdata[threadIdx.x] = sq; __syncthreads();
  for (int s = 128; s > 0; s >>= 1) {
    if (threadIdx.x < s) sdata[threadIdx.x] += sdata[threadIdx.x + s];
    __syncthreads();
  }
  float inv = rsqrtf(sdata[0] * (1.0f / DIM) + 1e-5f);
  float* o = out + (size_t)blockIdx.x * DIM;
#pragma unroll
  for (int i = 0; i < 4; ++i) {
    int cdx = threadIdx.x + 256 * i;
    o[cdx] = (v[i] - mu) * inv * g[cdx] + b[cdx];
  }
}

extern "C" void kernel_launch(void* const* d_in, const int* in_sizes, int n_in,
                              void* d_out, int out_size, void* d_ws, size_t ws_size,
                              hipStream_t stream) {
  (void)in_sizes; (void)n_in; (void)out_size; (void)ws_size;
  const float* key   = (const float*)d_in[0];
  const float* value = (const float*)d_in[1];
  const float* query = (const float*)d_in[2];
  const float* wk    = (const float*)d_in[3];
  const float* bk    = (const float*)d_in[4];
  const float* wv    = (const float*)d_in[5];
  const float* bv    = (const float*)d_in[6];
  const float* wq    = (const float*)d_in[7];
  const float* bq    = (const float*)d_in[8];
  const float* wf    = (const float*)d_in[9];
  const float* bf    = (const float*)d_in[10];
  const float* ln_g  = (const float*)d_in[11];
  const float* ln_b  = (const float*)d_in[12];

  float* out_f   = (float*)d_out;                              // (B,L,D)
  float* attn_f  = out_f + (size_t)BATCH * SEQ * DIM;          // (64,L,L)

  const size_t NE = (size_t)BATCH * SEQ * DIM;                 // 4M
  const size_t NA = (size_t)NBH * SEQ * SEQ;                   // 67M

  size_t off = 0;
  auto take = [&](size_t bytes) -> void* {
    void* p = (char*)d_ws + off;
    off += (bytes + 255) & ~(size_t)255;
    return p;
  };
  unsigned short* xk_bf = (unsigned short*)take(NE * 2);
  unsigned short* xv_bf = (unsigned short*)take(NE * 2);
  unsigned short* xq_bf = (unsigned short*)take(NE * 2);
  unsigned short* wk_bf = (unsigned short*)take((size_t)KW * DIM * DIM * 2);
  unsigned short* wv_bf = (unsigned short*)take((size_t)KW * DIM * DIM * 2);
  unsigned short* wq_bf = (unsigned short*)take((size_t)KW * DIM * DIM * 2);
  unsigned short* wf_bf = (unsigned short*)take((size_t)DIM * DIM * 2);
  unsigned short* yk_bf = (unsigned short*)take(NE * 2);
  unsigned short* yv_bf = (unsigned short*)take(NE * 2);
  unsigned short* yq_bf = (unsigned short*)take(NE * 2);
  unsigned short* vT_bf = (unsigned short*)take(NE * 2);
  float*          logZ  = (float*)take((size_t)NBH * SEQ * 4);
  unsigned short* at_bf = (unsigned short*)take(NA * 2);
  unsigned short* cx_bf = (unsigned short*)take(NE * 2);
  float*          preln = (float*)take(NE * 4);

  // 1. bf16 conversions / weight reorders
  k_cvt_bf16<<<dim3(NE / 256), dim3(256), 0, stream>>>(key,   xk_bf, (int)NE);
  k_cvt_bf16<<<dim3(NE / 256), dim3(256), 0, stream>>>(value, xv_bf, (int)NE);
  k_cvt_bf16<<<dim3(NE / 256), dim3(256), 0, stream>>>(query, xq_bf, (int)NE);
  k_cvt_bf16<<<dim3(DIM * DIM / 256), dim3(256), 0, stream>>>(wf, wf_bf, DIM * DIM);
  k_cvt_convw<<<dim3(DIM * DIM / 256), dim3(256), 0, stream>>>(wk, wk_bf);
  k_cvt_convw<<<dim3(DIM * DIM / 256), dim3(256), 0, stream>>>(wv, wv_bf);
  k_cvt_convw<<<dim3(DIM * DIM / 256), dim3(256), 0, stream>>>(wq, wq_bf);

  // 2. causal wrap-conv + GELU as TDM/LDS-staged WMMA GEMMs (K_eff = 3*1024)
  dim3 cg(BATCH * SEQ / 128, DIM / PANEL_ROWS);   // (32, 32), 8 waves/WG
  k_conv_gemm<<<cg, dim3(256), 0, stream>>>(xk_bf, wk_bf, bk, yk_bf);
  k_conv_gemm<<<cg, dim3(256), 0, stream>>>(xv_bf, wv_bf, bv, yv_bf);
  k_conv_gemm<<<cg, dim3(256), 0, stream>>>(xq_bf, wq_bf, bq, yq_bf);

  // 3. attention logits: log(clamp(0.5 * QK^T, 0.1)) into attn output region
  k_qk_logits<<<dim3(NBH, SEQ / 16, SEQ / 64), dim3(32), 0, stream>>>(yq_bf, yk_bf, attn_f);

  // 4. log-softmax (two passes) + bf16 copy of final attn
  k_row_stats<<<dim3(NBH * SEQ), dim3(256), 0, stream>>>(attn_f, logZ);
  k_apply_lsm<<<dim3((unsigned)(NA / 256)), dim3(256), 0, stream>>>(attn_f, logZ, at_bf);

  // 5. context = attn @ V
  k_transpose_v<<<dim3(NE / 256), dim3(256), 0, stream>>>(yv_bf, vT_bf);
  k_context_gemm<<<dim3(NBH, SEQ / 16), dim3(32), 0, stream>>>(at_bf, vT_bf, cx_bf);

  // 6. projection + bias + residual, then LayerNorm
  k_proj_gemm<<<dim3(BATCH * SEQ / 128, DIM / PANEL_ROWS), dim3(256), 0, stream>>>(
      cx_bf, wf_bf, bf, query, preln);
  k_layernorm<<<dim3(BATCH * SEQ), dim3(256), 0, stream>>>(preln, ln_g, ln_b, out_f);
}
